// StackLSTM_4913442586742
// MI455X (gfx1250) — matrix operand order repaired and checked by reading, not compiled, not run on hardware
//
#include <hip/hip_runtime.h>
#include <hip/hip_bf16.h>

// ---------------- problem constants (from reference) ----------------
#define T_STEPS 64
#define BATCH   128
#define HID     256
#define NLAYER  2
#define G4H     (4 * HID)        // 1024 gate columns (i,f,g,o)
#define KTOT    (2 * HID)        // 512 = [inp | h_prev] fused K
#define LDS_STRIDE (KTOT + 8)    // pad: keeps 16B alignment, staggers banks
#define NBLOCKS 16               // 8 M-tiles x 2 N-groups (persistent grid)

typedef __attribute__((ext_vector_type(16))) __bf16 v16bf;
typedef __attribute__((ext_vector_type(8)))  __bf16 v8bf;
typedef __attribute__((ext_vector_type(8)))  float  v8f;

__device__ __forceinline__ float sigf(float x) {
    return 1.0f / (1.0f + __expf(-x));
}
__device__ __forceinline__ float tanh_fast(float x) {
    float e = __expf(2.0f * x);
    return (e - 1.0f) / (e + 1.0f);
}

// ---------------- workspace layout (bytes) ----------------
#define OFF_WPACK 0
#define SZ_WPACK  ((size_t)NLAYER * G4H * KTOT * 2)
#define OFF_BSUM  (OFF_WPACK + SZ_WPACK)
#define SZ_BSUM   ((size_t)NLAYER * G4H * 4)
#define OFF_POS   (OFF_BSUM + SZ_BSUM)
#define SZ_POS    ((size_t)T_STEPS * BATCH * 4)
#define OFF_HST   (OFF_POS + SZ_POS)
#define SZ_STACK  ((size_t)(T_STEPS + 1) * BATCH * NLAYER * HID * 4)
#define OFF_CST   (OFF_HST + SZ_STACK)
#define OFF_HCUR  (OFF_CST + SZ_STACK)
#define SZ_HCUR   ((size_t)BATCH * HID * 4)
#define OFF_SYNC  (OFF_HCUR + SZ_HCUR)   // 2 x unsigned (counter, generation)

// ---------------- prep kernels ----------------
__global__ void pack_weights_kernel(const float* __restrict__ Wih,
                                    const float* __restrict__ Whh,
                                    __bf16* __restrict__ wpack) {
    size_t idx = (size_t)blockIdx.x * blockDim.x + threadIdx.x;  // L*4H*KTOT
    int l = (int)(idx / ((size_t)G4H * KTOT));
    size_t r = idx % ((size_t)G4H * KTOT);
    int n = (int)(r / KTOT);
    int k = (int)(r % KTOT);
    float v = (k < HID) ? Wih[((size_t)l * G4H + n) * HID + k]
                        : Whh[((size_t)l * G4H + n) * HID + (k - HID)];
    wpack[idx] = (__bf16)v;
}

__global__ void bias_kernel(const float* __restrict__ bih,
                            const float* __restrict__ bhh,
                            float* __restrict__ bsum) {
    int idx = blockIdx.x * blockDim.x + threadIdx.x;  // L*4H
    bsum[idx] = bih[idx] + bhh[idx];
}

__global__ void pos_kernel(const int* __restrict__ ops, int* __restrict__ posb) {
    int b = threadIdx.x;  // BATCH threads
    int p = 0;
    for (int t = 0; t < T_STEPS; ++t) {
        posb[t * BATCH + b] = p;
        p += ops[t * BATCH + b];
    }
}

__global__ void zero_slot0_kernel(float* __restrict__ hstack,
                                  float* __restrict__ cstack) {
    int idx = blockIdx.x * blockDim.x + threadIdx.x;  // B*L*H
    hstack[idx] = 0.0f;
    cstack[idx] = 0.0f;
}

__global__ void zero_sync_kernel(unsigned* __restrict__ sync) {
    sync[0] = 0u;  // arrival counter
    sync[1] = 0u;  // generation
}

// ---------------- device-scope grid barrier (16 co-resident blocks) ---------
__device__ __forceinline__ void grid_sync(unsigned* counter, unsigned* gen,
                                          unsigned nblocks) {
    __threadfence();   // flush this thread's stores to device scope (L2)
    __syncthreads();   // whole block done with the phase
    if (threadIdx.x == 0) {
        unsigned g = __hip_atomic_load(gen, __ATOMIC_RELAXED,
                                       __HIP_MEMORY_SCOPE_AGENT);
        unsigned a = __hip_atomic_fetch_add(counter, 1u, __ATOMIC_ACQ_REL,
                                            __HIP_MEMORY_SCOPE_AGENT);
        if (a + 1u == nblocks) {
            __hip_atomic_store(counter, 0u, __ATOMIC_RELAXED,
                               __HIP_MEMORY_SCOPE_AGENT);
            __hip_atomic_store(gen, g + 1u, __ATOMIC_RELEASE,
                               __HIP_MEMORY_SCOPE_AGENT);
        } else {
            while (__hip_atomic_load(gen, __ATOMIC_ACQUIRE,
                                     __HIP_MEMORY_SCOPE_AGENT) == g) {
                __builtin_amdgcn_s_sleep(2);
            }
        }
    }
    __syncthreads();
    __threadfence();   // acquire side: drop stale lines before next reads
}

// ---------------- one fused GEMM + LSTM-cell phase --------------------------
// Wave w in N-group owns 16-col tile n0 of the H=256 hidden columns and
// computes ALL FOUR gate tiles (i,f,g,o at n0, n0+H, n0+2H, n0+3H) so the
// LSTM pointwise is lane-local on the WMMA accumulators.
__device__ __forceinline__ void lstm_phase(
        const float* __restrict__ inp,     // (B,H): x_t or h of layer 0
        const __bf16* __restrict__ wpack,  // [4H][KTOT] this layer
        const float* __restrict__ bsum,    // [4H] this layer
        const int* __restrict__ post,      // [B] pos at step t
        float* __restrict__ hstack,        // [T+1][B][L][H]
        float* __restrict__ cstack,
        float* __restrict__ hout,          // (B,H): hcur or out_t
        int layer, int mTile, int nGroup,
        __bf16* aTile, int* posRow) {
    const int tid = threadIdx.x;

    // ---- stage A = bf16([inp | hstack[pos]]) into LDS (16 rows x 512) ----
    {
        const int row = tid >> 4;         // 0..15
        const int cg  = tid & 15;         // 32-col chunk id
        const int b   = mTile * 16 + row;
        const int p   = post[b];
        if (cg == 0) posRow[row] = p;
        __bf16* dst = &aTile[row * LDS_STRIDE + cg * 32];
        const float* src;
        if (cg < 8) {
            src = inp + (size_t)b * HID + cg * 32;
        } else {
            size_t base = ((size_t)p * BATCH + b) * (NLAYER * HID)
                        + (size_t)layer * HID + (cg - 8) * 32;
            src = hstack + base;
        }
        const float4* s4 = (const float4*)src;
#pragma unroll
        for (int c = 0; c < 8; ++c) {
            float4 v = s4[c];
            dst[4 * c + 0] = (__bf16)v.x;
            dst[4 * c + 1] = (__bf16)v.y;
            dst[4 * c + 2] = (__bf16)v.z;
            dst[4 * c + 3] = (__bf16)v.w;
        }
    }
    __syncthreads();

    const int wave = tid >> 5;
    const int lane = tid & 31;
    const int nl   = lane & 15;           // N col within tile / M row for A
    const int kh   = lane >> 4;           // K-half selector (contiguous k = 16*kh + s, this code's order)
    const int n0   = nGroup * 8 + wave;   // 16-col tile of H
    const int hcol = n0 * 16 + nl;        // 0..255 hidden column

    // A fragment: 16-bit A 16x32 layout (both lane halves M = lane&15;
    // lanes 0-15 take K {0..15}, lanes 16-31 take K {16..31}, matching B).
    const __bf16* aRow = &aTile[nl * LDS_STRIDE];

    // B fragment: lane&15 = N col, lane>>4 = K half of 16 contiguous bf16
    // -> one aligned 32B load per gate per k-step, streamed from L2.
    const __bf16* bptr[4];
#pragma unroll
    for (int g = 0; g < 4; ++g)
        bptr[g] = wpack + ((size_t)(g * HID + hcol)) * KTOT + kh * 16;

    v8f z = {0.f, 0.f, 0.f, 0.f, 0.f, 0.f, 0.f, 0.f};
    v8f acc[4];
    acc[0] = z; acc[1] = z; acc[2] = z; acc[3] = z;

#pragma unroll 4
    for (int kbase = 0; kbase < KTOT; kbase += 32) {
        v8bf a0 = *(const v8bf*)(aRow + kbase + kh * 16);
        v8bf a1 = *(const v8bf*)(aRow + kbase + kh * 16 + 8);
        v16bf a = __builtin_shufflevector(a0, a1,
            0, 1, 2, 3, 4, 5, 6, 7, 8, 9, 10, 11, 12, 13, 14, 15);
#pragma unroll
        for (int g = 0; g < 4; ++g) {
            v16bf bf = *(const v16bf*)(bptr[g] + kbase);
            if (kbase + 32 < KTOT)
                __builtin_prefetch(bptr[g] + kbase + 32, 0, 1);
            acc[g] = __builtin_amdgcn_wmma_f32_16x16x32_bf16(
                false, a, false, bf, (short)0, acc[g], false, false);
        }
    }

    // ---- lane-local LSTM pointwise on accumulators ----
    // C/D layout: VGPR v holds (M = v + 8*kh, N = lane&15).
    const float bi  = bsum[0 * HID + hcol];
    const float bf_ = bsum[1 * HID + hcol];
    const float bg  = bsum[2 * HID + hcol];
    const float bo  = bsum[3 * HID + hcol];

#pragma unroll
    for (int v = 0; v < 8; ++v) {
        const int m2 = v + 8 * kh;
        const int b  = mTile * 16 + m2;
        const int p  = posRow[m2];
        const size_t rd = ((size_t)p * BATCH + b) * (NLAYER * HID)
                        + (size_t)layer * HID + hcol;
        const size_t wr = ((size_t)(p + 1) * BATCH + b) * (NLAYER * HID)
                        + (size_t)layer * HID + hcol;
        const float cprev = cstack[rd];
        const float gi = sigf(acc[0][v] + bi);
        const float gf = sigf(acc[1][v] + bf_);
        const float gg = tanh_fast(acc[2][v] + bg);
        const float go = sigf(acc[3][v] + bo);
        const float c  = gf * cprev + gi * gg;
        const float h  = go * tanh_fast(c);
        cstack[wr] = c;
        hstack[wr] = h;
        hout[(size_t)b * HID + hcol] = h;
    }
}

// ---------------- persistent kernel: all 64 steps x 2 layers ---------------
__global__ __launch_bounds__(256)
void stacklstm_persistent_kernel(const float* __restrict__ x,
                                 const __bf16* __restrict__ wpack,
                                 const float* __restrict__ bsum,
                                 const int* __restrict__ posb,
                                 float* __restrict__ hstack,
                                 float* __restrict__ cstack,
                                 float* __restrict__ hcur,
                                 float* __restrict__ out,
                                 unsigned* __restrict__ sync) {
    __shared__ __align__(16) __bf16 aTile[16 * LDS_STRIDE];
    __shared__ int posRow[16];

    const int mTile  = blockIdx.x >> 1;   // 0..7
    const int nGroup = blockIdx.x & 1;    // 0..1
    unsigned* counter = sync;
    unsigned* gen     = sync + 1;
    const unsigned nb = gridDim.x;

#pragma unroll 1
    for (int t = 0; t < T_STEPS; ++t) {
        // layer 0: input = x_t, hidden -> hcur
        lstm_phase(x + (size_t)t * BATCH * HID, wpack, bsum,
                   posb + t * BATCH, hstack, cstack, hcur,
                   0, mTile, nGroup, aTile, posRow);
        grid_sync(counter, gen, nb);
        // layer 1: input = hcur, hidden -> output slice
        lstm_phase(hcur, wpack + (size_t)G4H * KTOT, bsum + G4H,
                   posb + t * BATCH, hstack, cstack,
                   out + (size_t)t * BATCH * HID,
                   1, mTile, nGroup, aTile, posRow);
        grid_sync(counter, gen, nb);
    }
}

// ---------------- host launcher ----------------
extern "C" void kernel_launch(void* const* d_in, const int* in_sizes, int n_in,
                              void* d_out, int out_size, void* d_ws, size_t ws_size,
                              hipStream_t stream) {
    (void)in_sizes; (void)n_in; (void)out_size; (void)ws_size;
    const float* x   = (const float*)d_in[0];   // (T,B,H)
    const int*   ops = (const int*)d_in[1];     // (T,B)
    const float* Wih = (const float*)d_in[2];   // (L,4H,H)
    const float* Whh = (const float*)d_in[3];   // (L,4H,H)
    const float* bih = (const float*)d_in[4];   // (L,4H)
    const float* bhh = (const float*)d_in[5];   // (L,4H)
    float* out = (float*)d_out;                 // (T,B,H)

    char* ws = (char*)d_ws;
    __bf16*   wpack  = (__bf16*)(ws + OFF_WPACK);
    float*    bsum   = (float*)(ws + OFF_BSUM);
    int*      posb   = (int*)(ws + OFF_POS);
    float*    hstack = (float*)(ws + OFF_HST);
    float*    cstack = (float*)(ws + OFF_CST);
    float*    hcur   = (float*)(ws + OFF_HCUR);
    unsigned* sync   = (unsigned*)(ws + OFF_SYNC);

    // Prep: bf16-pack fused [W_ih|W_hh], fuse biases, pos prefix sums,
    // zero stack slot 0 (only slot read before written) and barrier words.
    pack_weights_kernel<<<(int)((size_t)NLAYER * G4H * KTOT / 256), 256, 0, stream>>>(
        Wih, Whh, wpack);
    bias_kernel<<<(NLAYER * G4H) / 256, 256, 0, stream>>>(bih, bhh, bsum);
    pos_kernel<<<1, BATCH, 0, stream>>>(ops, posb);
    zero_slot0_kernel<<<(BATCH * NLAYER * HID) / 256, 256, 0, stream>>>(hstack, cstack);
    zero_sync_kernel<<<1, 1, 0, stream>>>(sync);

    // One persistent kernel replaces 128 dependent launches; 16 blocks are
    // trivially co-resident, synced by a device-scope generation barrier.
    stacklstm_persistent_kernel<<<NBLOCKS, 256, 0, stream>>>(
        x, wpack, bsum, posb, hstack, cstack, hcur, out, sync);
}
